// GraphAttentionLayer_5703716569149
// MI455X (gfx1250) — matrix operand order repaired and checked by reading, not compiled, not run on hardware
//
#include <hip/hip_runtime.h>
#include <hip/hip_bf16.h>

typedef __attribute__((ext_vector_type(16))) _Float16     v16h;
typedef __attribute__((ext_vector_type(8)))  _Float16     v8h;
typedef __attribute__((ext_vector_type(8)))  float        v8f;
typedef __attribute__((ext_vector_type(4)))  unsigned int v4u;
typedef __attribute__((ext_vector_type(8)))  int          v8i;
typedef __attribute__((ext_vector_type(4)))  int          v4i;

#define N_NODES 8192
#define IN_F    512
#define OUT_F   256
#define ALPHA   0.2f
#define NEG_INF -9.0e15f

// Workspace layout (bytes), all 16B-aligned:
#define HT_OFF 0u                 // hT  : [256][8192] f16  = 4,194,304 B
#define WT_OFF 4194304u           // Wt  : [256][512]  f16  =   262,144 B
#define S1_OFF 4456448u           // s1  : [8192] f32       =    32,768 B
#define S2_OFF 4489216u           // s2  : [8192] f32       =    32,768 B

// ---------------------------------------------------------------------------
// Kernel 0: build f16 transposed weights Wt[f][k] = W[k][f]; zero s1/s2.
// ---------------------------------------------------------------------------
__global__ void gat_prep(const float* __restrict__ W, _Float16* __restrict__ Wt,
                         float* __restrict__ s1, float* __restrict__ s2) {
    int idx = blockIdx.x * blockDim.x + threadIdx.x;
    if (idx < OUT_F * IN_F) {
        int f = idx >> 9;        // / 512
        int k = idx & 511;
        Wt[idx] = (_Float16)W[k * OUT_F + f];
    }
    if (idx < N_NODES) { s1[idx] = 0.0f; s2[idx] = 0.0f; }
}

// ---------------------------------------------------------------------------
// Kernel 1: Hᵀ = Wᵀ × Xᵀ via v_wmma_f32_16x16x32_f16.
//   Wave computes a 16(f) x 16(i) tile of Hᵀ over K = 512.
//   Also accumulates s1[i] = h[i]·a1, s2[i] = h[i]·a2 from the fp32 acc.
// ---------------------------------------------------------------------------
__global__ void __launch_bounds__(128)
gat_gemm_ht(const float* __restrict__ X, const _Float16* __restrict__ Wt,
            const float* __restrict__ a_vec, _Float16* __restrict__ hT,
            float* __restrict__ s1, float* __restrict__ s2) {
    const int t    = threadIdx.x;
    const int lane = t & 31, wv = t >> 5;
    const int n    = lane & 15;       // column index within tile (node i)
    const int kh   = lane >> 4;       // K-half selector
    const int f0   = (blockIdx.y * 4 + wv) * 16;
    const int i0   = blockIdx.x * 16;
    const int row  = i0 + n;

    v8f acc = {};
    const _Float16* wrow = Wt + (size_t)(f0 + n) * IN_F + kh * 16;  // A: m = n
    const float*    xrow = X  + (size_t)row * IN_F + kh * 16;       // B: contiguous K

    for (int k0 = 0; k0 < IN_F; k0 += 32) {
        v8h alo = *(const v8h*)(wrow + k0);
        v8h ahi = *(const v8h*)(wrow + k0 + 8);
        v16h af;
#pragma unroll
        for (int e = 0; e < 8; ++e) { af[e] = alo[e]; af[e + 8] = ahi[e]; }

        float4 x0 = *(const float4*)(xrow + k0);
        float4 x1 = *(const float4*)(xrow + k0 + 4);
        float4 x2 = *(const float4*)(xrow + k0 + 8);
        float4 x3 = *(const float4*)(xrow + k0 + 12);
        v16h bf;
        bf[0]=(_Float16)x0.x;  bf[1]=(_Float16)x0.y;  bf[2]=(_Float16)x0.z;  bf[3]=(_Float16)x0.w;
        bf[4]=(_Float16)x1.x;  bf[5]=(_Float16)x1.y;  bf[6]=(_Float16)x1.z;  bf[7]=(_Float16)x1.w;
        bf[8]=(_Float16)x2.x;  bf[9]=(_Float16)x2.y;  bf[10]=(_Float16)x2.z; bf[11]=(_Float16)x2.w;
        bf[12]=(_Float16)x3.x; bf[13]=(_Float16)x3.y; bf[14]=(_Float16)x3.z; bf[15]=(_Float16)x3.w;

        acc = __builtin_amdgcn_wmma_f32_16x16x32_f16(false, af, false, bf,
                                                     (short)0, acc, false, false);
    }

    float p1 = 0.0f, p2 = 0.0f;
#pragma unroll
    for (int r = 0; r < 8; ++r) {
        int   f = f0 + kh * 8 + r;
        float v = acc[r];
        hT[(size_t)f * N_NODES + row] = (_Float16)v;
        p1 += v * a_vec[f];
        p2 += v * a_vec[OUT_F + f];
    }
    atomicAdd(&s1[row], p1);
    atomicAdd(&s2[row], p2);
}

// ---------------------------------------------------------------------------
// TDM descriptor: load a 16(row) x 128(col) int32 tile of adj into LDS,
// padding +4 DWORDs every 128 DWORDs (row) => LDS row stride 132 DWORDs
// so the 16-lane ds_read_b128 consumption pattern is bank-conflict-free.
// D# layout per CDNA5 ISA 8.3/8.4 (group0: count/lds/global+type,
// group1: data_size|pad cfg, tensor dims, tile dims, stride).
// ---------------------------------------------------------------------------
__device__ __forceinline__ void tdm_load_adj_tile(const int* gsrc, unsigned int lds_off) {
    unsigned long long ga = (unsigned long long)(size_t)gsrc;
    v4u g0;
    g0[0] = 1u;                                              // count=1, user desc
    g0[1] = lds_off;                                         // lds_addr (bytes)
    g0[2] = (unsigned int)ga;                                // global_addr[31:0]
    g0[3] = (unsigned int)((ga >> 32) & 0x1FFFFFFu) | (2u << 30); // addr[56:32] | type=2
    v8i g1;
    g1[0] = (int)((2u << 16) | (1u << 20) | (6u << 22) | (3u << 25)); // 4B elems, pad 4dw/128dw
    g1[1] = (int)(8192u << 16);      // tensor_dim0 = 8192 (bits 63:48 hold [15:0])
    g1[2] = (int)(8192u << 16);      // tensor_dim0[31:16]=0 | tensor_dim1[15:0]=8192
    g1[3] = (int)(128u  << 16);      // tensor_dim1[31:16]=0 | tile_dim0 = 128
    g1[4] = 16;                      // tile_dim1 = 16, tile_dim2 = 0
    g1[5] = 8192;                    // tensor_dim0_stride[31:0]
    g1[6] = 0;                       // stride0[47:32] | stride1[15:0]
    g1[7] = 0;
    v4i z4 = {0, 0, 0, 0};
    v8i z8 = {0, 0, 0, 0, 0, 0, 0, 0};
    // amdgpu-toolchain (clang-23) 6-arg form: (g0, g1, g2, g3, g4, cpol)
    __builtin_amdgcn_tensor_load_to_lds(g0, g1, z4, z4, z8, 0);
}

// ---------------------------------------------------------------------------
// Kernel 2: single-pass flash-style masked softmax + (attention @ h) + ELU.
//   One workgroup (128 thr / 4 waves) per 16-row tile; 128 columns per step.
//   adj tiles are double-buffered into LDS by the Tensor Data Mover
//   (tensor_load_to_lds + s_wait_tensorcnt), issued by wave 0 one block
//   ahead. Each wave computes ONE 32-column Pᵀ fragment, stages it in LDS;
//   all 4 waves consume all 4 fragments. Online-max rescale is a per-lane
//   scalar because the accumulator column == node row i.
// ---------------------------------------------------------------------------
__global__ void __launch_bounds__(128)
gat_attn(const int* __restrict__ adj, const _Float16* __restrict__ hT,
         const float* __restrict__ s1, const float* __restrict__ s2,
         float* __restrict__ out) {
    __shared__ __align__(16) int Ast[2][16][132];        // TDM-staged adj tiles (16.5 KB)
    __shared__ __align__(16) _Float16 Pst[4][32][16];    // staged Pᵀ fragments (4 KB)
    __shared__ float sm_m[16], sm_s[16], sm_newm[16], sm_c[16];
    __shared__ float cmax[4][16], csum[4][16];

    const int t    = threadIdx.x;
    const int lane = t & 31, wv = t >> 5;
    const int n    = lane & 15;        // node column i within tile
    const int kh   = lane >> 4;
    const int i0   = blockIdx.x * 16;
    const int iRow = i0 + n;

    if (t < 16) { sm_m[t] = -3.0e38f; sm_s[t] = 0.0f; }
    const float s1v = s1[iRow];

    v8f vz = {};
    v8f acc[4];
#pragma unroll
    for (int q = 0; q < 4; ++q) acc[q] = vz;

    const float4* s24 = (const float4*)s2;
    const int* tileBase = adj + (size_t)i0 * N_NODES;    // tile (i0..i0+15) x (j0..j0+127)
    const unsigned int ldsA0 = (unsigned int)(size_t)&Ast[0][0][0];
    const unsigned int ldsA1 = (unsigned int)(size_t)&Ast[1][0][0];

    // Prologue: DMA tile j0=0 into buffer 0 (wave 0 issues; EXEC-independent op)
    if (wv == 0) tdm_load_adj_tile(tileBase, ldsA0);
    __syncthreads();

    for (int j0 = 0; j0 < N_NODES; j0 += 128) {
        const int buf = (j0 >> 7) & 1;
        // ---- (0) prefetch next adj tile via TDM; wait for current ----
        if (wv == 0) {
            if (j0 + 128 < N_NODES) {
                tdm_load_adj_tile(tileBase + (j0 + 128), buf ? ldsA0 : ldsA1);
                __builtin_amdgcn_s_wait_tensorcnt(1);    // in-order: current tile done
            } else {
                __builtin_amdgcn_s_wait_tensorcnt(0);
            }
        }
        __syncthreads();

        // ---- (1) this wave's 32-column subchunk: masked leaky logits ----
        const int jb = wv * 32;                          // offset within the 128-tile
        const int4* arow = (const int4*)&Ast[buf][n][jb + kh * 16];
        const int   q0   = (j0 + jb + kh * 16) >> 2;
        float ev[16];
        float lmax = -3.0e38f;
#pragma unroll
        for (int blk = 0; blk < 4; ++blk) {
            int4   a4 = arow[blk];                       // ds_read_b128, conflict-free
            float4 z4 = s24[q0 + blk];
            float e0 = s1v + z4.x; e0 = e0 > 0.f ? e0 : ALPHA * e0; e0 = a4.x > 0 ? e0 : NEG_INF;
            float e1 = s1v + z4.y; e1 = e1 > 0.f ? e1 : ALPHA * e1; e1 = a4.y > 0 ? e1 : NEG_INF;
            float e2 = s1v + z4.z; e2 = e2 > 0.f ? e2 : ALPHA * e2; e2 = a4.z > 0 ? e2 : NEG_INF;
            float e3 = s1v + z4.w; e3 = e3 > 0.f ? e3 : ALPHA * e3; e3 = a4.w > 0 ? e3 : NEG_INF;
            ev[blk * 4 + 0] = e0; ev[blk * 4 + 1] = e1;
            ev[blk * 4 + 2] = e2; ev[blk * 4 + 3] = e3;
            lmax = fmaxf(lmax, fmaxf(fmaxf(e0, e1), fmaxf(e2, e3)));
        }
        // merge the two half-lanes that share column i (wave32 xor-16)
        float m2 = fmaxf(lmax, __shfl_xor(lmax, 16, 32));
        if (lane < 16) cmax[wv][lane] = m2;
        __syncthreads();

        // ---- (2) fold block max, compute per-column rescale factor ----
        if (t < 16) {
            float bm = sm_m[t];
            bm = fmaxf(bm, cmax[0][t]); bm = fmaxf(bm, cmax[1][t]);
            bm = fmaxf(bm, cmax[2][t]); bm = fmaxf(bm, cmax[3][t]);
            sm_newm[t] = bm;
            sm_c[t]    = __expf(sm_m[t] - bm);
            sm_m[t]    = bm;
        }
        __syncthreads();

        // ---- (3) exp weights -> LDS stage; chunk sums ----
        const float newm = sm_newm[n];
        float lsum = 0.0f;
        v8h plo, phi;
#pragma unroll
        for (int e = 0; e < 8; ++e) {
            float p0 = __expf(ev[e] - newm);
            float p1 = __expf(ev[e + 8] - newm);
            lsum += p0 + p1;
            plo[e] = (_Float16)p0;
            phi[e] = (_Float16)p1;
        }
        {
            v8h* pp = (v8h*)&Pst[wv][lane][0];
            pp[0] = plo; pp[1] = phi;                 // 2x ds_store_b128
        }
        float psum = lsum + __shfl_xor(lsum, 16, 32);
        if (lane < 16) csum[wv][lane] = psum;
        __syncthreads();

        // ---- (4) rescale accumulators (per-lane scalar) + 16 WMMAs ----
        const float c = sm_c[n];
#pragma unroll
        for (int q = 0; q < 4; ++q) acc[q] = acc[q] * c;
        if (t < 16)
            sm_s[t] = sm_s[t] * sm_c[t] +
                      (csum[0][t] + csum[1][t] + csum[2][t] + csum[3][t]);

#pragma unroll
        for (int fr = 0; fr < 4; ++fr) {
            const v8h* pp = (const v8h*)&Pst[fr][lane][0];
            v8h qlo = pp[0], qhi = pp[1];             // 2x ds_read_b128
            v16h pf;
#pragma unroll
            for (int e = 0; e < 8; ++e) { pf[e] = qlo[e]; pf[e + 8] = qhi[e]; }
            const int jf = j0 + fr * 32;
#pragma unroll
            for (int q = 0; q < 4; ++q) {
                const int f = (wv * 4 + q) * 16 + n;  // A row m = lane&15
                const _Float16* hrow = hT + (size_t)f * N_NODES + jf + kh * 16;
                v8h alo = *(const v8h*)(hrow);
                v8h ahi = *(const v8h*)(hrow + 8);
                v16h af;
#pragma unroll
                for (int e = 0; e < 8; ++e) { af[e] = alo[e]; af[e + 8] = ahi[e]; }
                acc[q] = __builtin_amdgcn_wmma_f32_16x16x32_f16(false, af, false, pf,
                                                                (short)0, acc[q], false, false);
            }
        }
        __syncthreads();   // protect Pst/sm_c and the adj buffer the next TDM overwrites
    }

    // ---- Epilogue: 1/rowsum, ELU, contiguous float4 stores ----
    const float inv = 1.0f / sm_s[n];
#pragma unroll
    for (int q = 0; q < 4; ++q) {
        const int fbase = (wv * 4 + q) * 16 + kh * 8;
        float v0[8];
#pragma unroll
        for (int r = 0; r < 8; ++r) {
            float v = acc[q][r] * inv;
            v0[r] = v > 0.0f ? v : (__expf(v) - 1.0f);
        }
        float4* dst = (float4*)(out + (size_t)iRow * OUT_F + fbase);
        dst[0] = make_float4(v0[0], v0[1], v0[2], v0[3]);
        dst[1] = make_float4(v0[4], v0[5], v0[6], v0[7]);
    }
}

// ---------------------------------------------------------------------------
extern "C" void kernel_launch(void* const* d_in, const int* in_sizes, int n_in,
                              void* d_out, int out_size, void* d_ws, size_t ws_size,
                              hipStream_t stream) {
    const float* X   = (const float*)d_in[0];   // [8192,512] f32
    const int*   adj = (const int*)d_in[1];     // [8192,8192] i32
    const float* W   = (const float*)d_in[2];   // [512,256] f32
    const float* av  = (const float*)d_in[3];   // [512,1] f32
    float* out = (float*)d_out;                 // [8192,256] f32

    char* ws = (char*)d_ws;
    _Float16* hT = (_Float16*)(ws + HT_OFF);
    _Float16* Wt = (_Float16*)(ws + WT_OFF);
    float*    s1 = (float*)(ws + S1_OFF);
    float*    s2 = (float*)(ws + S2_OFF);

    gat_prep<<<512, 256, 0, stream>>>(W, Wt, s1, s2);
    gat_gemm_ht<<<dim3(512, 4), 128, 0, stream>>>(X, Wt, av, hT, s1, s2);
    gat_attn<<<512, 128, 0, stream>>>(adj, hT, s1, s2, out);
}
